// ACMix_61083024883996
// MI455X (gfx1250) — matrix operand order repaired and checked by reading, not compiled, not run on hardware
//
#include <hip/hip_runtime.h>
#include <hip/hip_bf16.h>
#include <math.h>

// ---------------------------------------------------------------------------
// ACmix-style block for MI455X (gfx1250), wave32 + WMMA + TDM.
//   K1: q/k/v 1x1 projections as WMMA GEMMs; operand staging via
//       tensor_load_to_lds (TDM) with LDS padding for bank-conflict-free
//       fragment gathers; f32 tiles converted to f16 during fragment build.
//   K2: positional-encoding conv (analytic coordinate grid)
//   K3: fused 7x7 local attention + depthwise conv + BN + ReLU + blend
// ---------------------------------------------------------------------------

typedef __attribute__((ext_vector_type(16))) _Float16     v16h;
typedef __attribute__((ext_vector_type(8)))  float        v8f;
typedef __attribute__((ext_vector_type(4)))  unsigned int v4u;
typedef __attribute__((ext_vector_type(8)))  int          v8i_;
typedef __attribute__((ext_vector_type(4)))  int          v4i_;

#define B_    4
#define C_    64
#define HEADS 4
#define D_    16
#define H_    56
#define W_    56
#define HW_   (H_ * W_)
#define KA_   7
#define PAD_  3
#define TH_   4          // rows per block in the attention kernel
#define LDSW_ 64         // padded float row width in K3
#define ROWP_ 66         // padded LDS row stride (dwords) produced by TDM pad

// ---------------------------------------------------------------------------
// TDM: issue one 2D tile DMA (global f32 -> LDS) with 2-dword padding every
// 64 dwords (row stride becomes ROWP_=66 dwords in LDS).
// D# packing per CDNA5 ISA §8.3 (group0) / §8.4 (group1); groups 2/3 zero
// (2D tensor: tile_dim2 = 0).
// ---------------------------------------------------------------------------
__device__ __forceinline__ void tdm_load_2d_f32(
    unsigned lds_addr, const float* gptr,
    unsigned tdim0, unsigned tdim1,          // tensor extents (elements)
    unsigned tile0, unsigned tile1,          // tile extents (elements)
    unsigned long long stride0)              // dim0 stride (elements)
{
    const unsigned long long ga = (unsigned long long)(const void*)gptr;
    v4u g0;
    g0[0] = 1u;                                             // count=1 (valid)
    g0[1] = lds_addr;                                       // lds_addr[31:0]
    g0[2] = (unsigned)(ga & 0xFFFFFFFFull);                 // global_addr[31:0]
    g0[3] = ((unsigned)(ga >> 32) & 0x01FFFFFFu)            // global_addr[56:32]
          | (2u << 30);                                     // type = 2 (image)
    v8i_ g1;
    g1[0] = (int)((2u << 16)      // data_size = 4B
                | (1u << 20)      // pad_enable
                | (5u << 22)      // pad_interval: code 5 = every 64 dwords
                | (1u << 25));    // pad_amount:   code 1 = 2 dwords
    g1[1] = (int)((tdim0 & 0xFFFFu) << 16);                       // tensor_dim0[15:0]
    g1[2] = (int)(((tdim0 >> 16) & 0xFFFFu) |
                  ((tdim1 & 0xFFFFu) << 16));                     // dim0 hi | dim1 lo
    g1[3] = (int)(((tdim1 >> 16) & 0xFFFFu) | (tile0 << 16));     // dim1 hi | tile_dim0
    g1[4] = (int)(tile1 & 0xFFFFu);                               // tile_dim1 (tile_dim2=0)
    g1[5] = (int)(unsigned)(stride0 & 0xFFFFFFFFull);             // dim0_stride lo32
    g1[6] = (int)(unsigned)((stride0 >> 32) & 0xFFFFull);         // dim0_stride hi16
    g1[7] = 0;
    v4i_ z4 = {0, 0, 0, 0};
#if __clang_major__ >= 23
    v8i_ z8 = {0, 0, 0, 0, 0, 0, 0, 0};
    __builtin_amdgcn_tensor_load_to_lds(g0, g1, z4, z4, z8, 0);
#else
    __builtin_amdgcn_tensor_load_to_lds(g0, g1, z4, z4, 0);
#endif
}

// ============================================================================
// Kernel 1: q/k/v projections via v_wmma_f32_16x16x32_f16, TDM-staged tiles.
// Grid: (HW/64, B). Block: 128 threads (4 waves). Each wave owns 16 spatial
// columns (N); 3 projections x 4 output-channel tiles (M) x 2 K-chunks.
// ============================================================================
__global__ __launch_bounds__(128) void qkv_wmma_kernel(
    const float* __restrict__ x,
    const float* __restrict__ w1, const float* __restrict__ b1,
    const float* __restrict__ w2, const float* __restrict__ b2,
    const float* __restrict__ w3, const float* __restrict__ b3,
    float* __restrict__ qo, float* __restrict__ ko, float* __restrict__ vo)
{
    // f32 tiles with TDM-generated 66-dword row stride
    __shared__ float sWf[3][64 * ROWP_];   // [proj][o * 66 + c]   (~50KB)
    __shared__ float sXf[64 * ROWP_];      // [c * 66 + hw_local]  (~17KB)

    const int tid = threadIdx.x;
    const int b   = blockIdx.y;
    const int hw0 = blockIdx.x * 64;

    if (tid < 32) {  // wave 0 issues the DMAs (EXEC is ignored by TDM)
        tdm_load_2d_f32((unsigned)(size_t)(void*)&sWf[0][0], w1, 64, 64, 64, 64, 64);
        tdm_load_2d_f32((unsigned)(size_t)(void*)&sWf[1][0], w2, 64, 64, 64, 64, 64);
        tdm_load_2d_f32((unsigned)(size_t)(void*)&sWf[2][0], w3, 64, 64, 64, 64, 64);
        tdm_load_2d_f32((unsigned)(size_t)(void*)&sXf[0],
                        x + (size_t)b * C_ * HW_ + hw0, HW_, 64, 64, 64, HW_);
        __builtin_amdgcn_s_wait_tensorcnt(0);
    }
    __syncthreads();

    const int  wid  = tid >> 5;
    const int  lane = tid & 31;
    const int  nrow = lane & 15;
    const bool hi   = lane >= 16;
    const int  colN = wid * 16 + nrow;   // spatial column owned by this lane

    // B fragments (K=0..31 and K=32..63). contiguous k order used here:
    // lanes 0-15 hold K=0..15, lanes 16-31 hold K=16..31 of the chunk.
    v16h bf0, bf1;
    {
        const int kofs = hi ? 16 : 0;
        #pragma unroll
        for (int h = 0; h < 16; ++h) {
            bf0[h] = (_Float16)sXf[(kofs + h) * ROWP_ + colN];
            bf1[h] = (_Float16)sXf[(32 + kofs + h) * ROWP_ + colN];
        }
    }

    const float* bias[3] = { b1, b2, b3 };
    float*       outp[3] = { qo, ko, vo };

    #pragma unroll
    for (int p = 0; p < 3; ++p) {
        float* op = outp[p];
        const float* bp_ = bias[p];
        #pragma unroll
        for (int m = 0; m < 4; ++m) {
            // A fragments (16x32), k order matching B:
            // lanes 0-15:  halves 0..15 -> K 0..15
            // lanes 16-31: halves 0..15 -> K 16..31
            v16h a0, a1;
            #pragma unroll
            for (int h = 0; h < 16; ++h) {
                const int k0  = h + (hi ? 16 : 0);
                a0[h] = (_Float16)sWf[p][(m * 16 + nrow) * ROWP_ + k0];
                a1[h] = (_Float16)sWf[p][(m * 16 + nrow) * ROWP_ + 32 + k0];
            }
            v8f acc = {};
            acc = __builtin_amdgcn_wmma_f32_16x16x32_f16(
                      false, a0, false, bf0, (short)0, acc, false, false);
            acc = __builtin_amdgcn_wmma_f32_16x16x32_f16(
                      false, a1, false, bf1, (short)0, acc, false, false);
            // C layout: VGPR r -> row m*16 + r (+8 for upper half-wave), N = lane&15.
            #pragma unroll
            for (int r = 0; r < 8; ++r) {
                const int o = m * 16 + r + (hi ? 8 : 0);
                op[((size_t)b * C_ + o) * HW_ + hw0 + colN] = acc[r] + bp_[o];
            }
        }
    }
}

// ============================================================================
// Kernel 2: positional encoding. pe[d,h,w] = bp[d] + conv3x3(zero-pad) over
// the analytic (loc_w, loc_h) coordinate planes.
// ============================================================================
__global__ __launch_bounds__(256) void pe_kernel(
    const float* __restrict__ wp, const float* __restrict__ bp,
    float* __restrict__ pe)
{
    const int i = blockIdx.x * 256 + threadIdx.x;
    if (i >= D_ * HW_) return;
    const int d  = i / HW_;
    const int hw = i % HW_;
    const int h  = hw / W_, w = hw % W_;

    float acc = bp[d];
    #pragma unroll
    for (int kh = 0; kh < 3; ++kh) {
        #pragma unroll
        for (int kw = 0; kw < 3; ++kw) {
            const int hh = h + kh - 1, ww = w + kw - 1;
            if (hh >= 0 && hh < H_ && ww >= 0 && ww < W_) {
                const float lw = -1.0f + (2.0f / (float)(W_ - 1)) * (float)ww;
                const float lh = -1.0f + (2.0f / (float)(H_ - 1)) * (float)hh;
                acc += wp[(d * 2 + 0) * 9 + kh * 3 + kw] * lw
                     + wp[(d * 2 + 1) * 9 + kh * 3 + kw] * lh;
            }
        }
    }
    pe[i] = acc;
}

// ============================================================================
// Kernel 3: fused local attention + depthwise conv + BN + ReLU + blend.
// Grid: (H/TH_, B*HEADS). Block: 224 threads (7 waves), one thread per pixel
// of a 4x56 strip. (k+pe) is summed before the window gather: both receive
// the identical reflect padding so the sum commutes with unfold.
// ============================================================================
__global__ __launch_bounds__(224) void attn_conv_kernel(
    const float* __restrict__ x,
    const float* __restrict__ qw, const float* __restrict__ kw_,
    const float* __restrict__ vw, const float* __restrict__ pe,
    const float* __restrict__ w_dw, const float* __restrict__ b_dw,
    const float* __restrict__ gamma, const float* __restrict__ beta,
    const float* __restrict__ mean,  const float* __restrict__ var,
    const float* __restrict__ rate1, const float* __restrict__ rate2,
    float* __restrict__ out)
{
    __shared__ float sKP[D_][TH_ + 6][LDSW_];   // k + pe, reflect halo (40KB)
    __shared__ float sV [D_][TH_ + 6][LDSW_];   // v, reflect halo       (40KB)
    __shared__ float sX [D_][TH_ + 2][LDSW_];   // x, zero halo          (24KB)

    const int tid  = threadIdx.x;
    const int r0   = blockIdx.x * TH_;
    const int bh   = blockIdx.y;
    const int b    = bh >> 2, head = bh & 3;
    const size_t cbase = ((size_t)b * C_ + head * D_) * HW_;

    // --- stage attention tiles with reflect padding ---
    const int NKP = D_ * (TH_ + 6) * (W_ + 6);
    for (int i = tid; i < NKP; i += 224) {
        const int d   = i / ((TH_ + 6) * (W_ + 6));
        const int rem = i % ((TH_ + 6) * (W_ + 6));
        const int rr  = rem / (W_ + 6), cc = rem % (W_ + 6);
        int gr = r0 + rr - PAD_;
        gr = (gr < 0) ? -gr : ((gr >= H_) ? 2 * H_ - 2 - gr : gr);
        int gc = cc - PAD_;
        gc = (gc < 0) ? -gc : ((gc >= W_) ? 2 * W_ - 2 - gc : gc);
        const size_t gidx = cbase + (size_t)d * HW_ + gr * W_ + gc;
        sKP[d][rr][cc] = kw_[gidx] + pe[(d * H_ + gr) * W_ + gc];
        sV [d][rr][cc] = vw[gidx];
    }
    // --- stage x tile with zero padding (depthwise conv) ---
    const int NX = D_ * (TH_ + 2) * (W_ + 2);
    for (int i = tid; i < NX; i += 224) {
        const int d   = i / ((TH_ + 2) * (W_ + 2));
        const int rem = i % ((TH_ + 2) * (W_ + 2));
        const int rr  = rem / (W_ + 2), cc = rem % (W_ + 2);
        const int gr = r0 + rr - 1, gc = cc - 1;
        float v = 0.0f;
        if (gr >= 0 && gr < H_ && gc >= 0 && gc < W_)
            v = x[cbase + (size_t)d * HW_ + gr * W_ + gc];
        sX[d][rr][cc] = v;
    }
    __syncthreads();

    const int y  = tid / W_;
    const int xq = tid % W_;
    const int gy = r0 + y;

    // q for this pixel, pre-scaled by D^-0.5 = 0.25
    float qd[D_];
    #pragma unroll
    for (int d = 0; d < D_; ++d)
        qd[d] = qw[cbase + (size_t)d * HW_ + gy * W_ + xq] * 0.25f;

    // logits over the 7x7 window
    float att[KA_ * KA_];
    float mx = -3.4e38f;
    #pragma unroll
    for (int kh = 0; kh < KA_; ++kh) {
        #pragma unroll
        for (int kq = 0; kq < KA_; ++kq) {
            float s = 0.0f;
            #pragma unroll
            for (int d = 0; d < D_; ++d)
                s += qd[d] * sKP[d][y + kh][xq + kq];
            att[kh * KA_ + kq] = s;
            mx = fmaxf(mx, s);
        }
    }
    float den = 0.0f;
    #pragma unroll
    for (int i = 0; i < KA_ * KA_; ++i) { att[i] = __expf(att[i] - mx); den += att[i]; }
    const float rden = 1.0f / den;
    #pragma unroll
    for (int i = 0; i < KA_ * KA_; ++i) att[i] *= rden;

    const float r1 = rate1[0], r2 = rate2[0];
    #pragma unroll
    for (int d = 0; d < D_; ++d) {
        // attention output
        float acc = 0.0f;
        #pragma unroll
        for (int kh = 0; kh < KA_; ++kh) {
            #pragma unroll
            for (int kq = 0; kq < KA_; ++kq)
                acc += att[kh * KA_ + kq] * sV[d][y + kh][xq + kq];
        }
        // depthwise conv + BN(eval) + ReLU
        const int c = head * D_ + d;
        float cv = b_dw[c];
        #pragma unroll
        for (int kh = 0; kh < 3; ++kh) {
            #pragma unroll
            for (int kq = 0; kq < 3; ++kq)
                cv += w_dw[c * 9 + kh * 3 + kq] * sX[d][y + kh][xq + kq];
        }
        const float bns = gamma[c] * rsqrtf(var[c] + 1e-5f);
        float yv = (cv - mean[c]) * bns + beta[c];
        yv = fmaxf(yv, 0.0f);

        out[cbase + (size_t)d * HW_ + gy * W_ + xq] = r1 * acc + r2 * yv;
    }
}

// ============================================================================
extern "C" void kernel_launch(void* const* d_in, const int* in_sizes, int n_in,
                              void* d_out, int out_size, void* d_ws, size_t ws_size,
                              hipStream_t stream) {
    const float* x     = (const float*)d_in[0];
    const float* w1    = (const float*)d_in[1];
    const float* b1    = (const float*)d_in[2];
    const float* w2    = (const float*)d_in[3];
    const float* b2    = (const float*)d_in[4];
    const float* w3    = (const float*)d_in[5];
    const float* b3    = (const float*)d_in[6];
    const float* wp    = (const float*)d_in[7];
    const float* bp    = (const float*)d_in[8];
    const float* w_dw  = (const float*)d_in[9];
    const float* b_dw  = (const float*)d_in[10];
    const float* gamma = (const float*)d_in[11];
    const float* beta  = (const float*)d_in[12];
    const float* mean  = (const float*)d_in[13];
    const float* var   = (const float*)d_in[14];
    const float* rate1 = (const float*)d_in[15];
    const float* rate2 = (const float*)d_in[16];
    float* out = (float*)d_out;

    // Workspace: q, k, v [B,C,H,W] f32 + pe [D,H,W] f32  (~9.8 MB)
    float* qw = (float*)d_ws;
    float* kw = qw + (size_t)B_ * C_ * HW_;
    float* vw = kw + (size_t)B_ * C_ * HW_;
    float* pe = vw + (size_t)B_ * C_ * HW_;

    qkv_wmma_kernel<<<dim3(HW_ / 64, B_), 128, 0, stream>>>(
        x, w1, b1, w2, b2, w3, b3, qw, kw, vw);
    pe_kernel<<<(D_ * HW_ + 255) / 256, 256, 0, stream>>>(wp, bp, pe);
    attn_conv_kernel<<<dim3(H_ / TH_, B_ * HEADS), 224, 0, stream>>>(
        x, qw, kw, vw, pe, w_dw, b_dw, gamma, beta, mean, var, rate1, rate2, out);
}
